// RNNSurrogate_20272245637292
// MI455X (gfx1250) — matrix operand order repaired and checked by reading, not compiled, not run on hardware
//
#include <hip/hip_runtime.h>
#include <hip/hip_bf16.h>

typedef _Float16 half8  __attribute__((ext_vector_type(8)));
typedef _Float16 half16 __attribute__((ext_vector_type(16)));
typedef float    floatx8 __attribute__((ext_vector_type(8)));

#define B_    256
#define T_    365
#define TSEQ  73
#define REP_  5
#define DX    128
#define DS    32
#define UNITS 512
#define NOUT  64
#define KTOT  672   /* DX + DS + UNITS */
#define NZ    2048  /* 4*UNITS */
#define NKC   (KTOT / 32)   /* 21 k-chunks */

#if defined(__has_builtin)
#  if __has_builtin(__builtin_amdgcn_sched_barrier)
#    define SCHED_FENCE() __builtin_amdgcn_sched_barrier(0)
#  endif
#endif
#ifndef SCHED_FENCE
#  define SCHED_FENCE() ((void)0)
#endif

union Frag16 { half16 v; half8 h[2]; };

__device__ __forceinline__ float sigf(float x) { return 1.0f / (1.0f + __expf(-x)); }

// A-matrix source for k-chunk starting at kg (compile-time constant under full unroll)
__device__ __forceinline__ const _Float16* a_src(int kg, int row,
        const _Float16* __restrict__ xv, const _Float16* __restrict__ xsp,
        const _Float16* __restrict__ hin) {
    if (kg < DX)           return xv  + row * DX    + kg;
    if (kg < DX + DS)      return xsp + row * DS    + (kg - DX);
    return hin + row * UNITS + (kg - (DX + DS));
}

// ---------------- precompute kernels (run once per call) ----------------

__global__ void prep_static_k(const float* __restrict__ x, const float* __restrict__ xm,
                              const float* __restrict__ xs, _Float16* __restrict__ xv) {
    int i = blockIdx.x * 256 + threadIdx.x;      // B_*DX
    int d = i % DX;
    xv[i] = (_Float16)((x[i] - xm[d]) / xs[d]);
}

__global__ void prep_seq_k(const float* __restrict__ xseq, const float* __restrict__ m,
                           const float* __restrict__ s, _Float16* __restrict__ out) {
    int i = blockIdx.x * 256 + threadIdx.x;      // TSEQ*B_*DS, layout [ts][b][ds]
    int ts = i / (B_ * DS);
    int r  = i % (B_ * DS);
    int b  = r / DS;
    int d  = r % DS;
    out[i] = (_Float16)((xseq[(b * TSEQ + ts) * DS + d] - m[d]) / s[d]);
}

// WcatT[n][k] = (k<160 ? Wx[k][n] : Wh[k-160][n]) as f16  -> pre-transposed B matrix
__global__ void prep_wcat_k(const float* __restrict__ Wx, const float* __restrict__ Wh,
                            _Float16* __restrict__ WcatT) {
    int i = blockIdx.x * 256 + threadIdx.x;      // NZ*KTOT
    int n = i / KTOT;
    int k = i % KTOT;
    float v = (k < DX + DS) ? Wx[k * NZ + n] : Wh[(k - (DX + DS)) * NZ + n];
    WcatT[i] = (_Float16)v;
}

__global__ void prep_wd_k(const float* __restrict__ Wd, _Float16* __restrict__ WdT) {
    int i = blockIdx.x * 256 + threadIdx.x;      // NOUT*UNITS, WdT[n][k]
    int n = i / UNITS;
    int k = i % UNITS;
    WdT[i] = (_Float16)Wd[k * NOUT + n];
}

__global__ void init_state_k(float* __restrict__ c, _Float16* __restrict__ h) {
    int i = blockIdx.x * 256 + threadIdx.x;      // B_*UNITS
    c[i] = 0.0f;
    h[i] = (_Float16)0.0f;
}

// ---------------- fused LSTM step ----------------
// grid (4, 8): blockIdx.x = 64-row batch block, blockIdx.y = 64-unit block.
// Wave tile: 32 rows x 64 z-cols (2 M-tiles x 4 N-tiles); wave>>1 selects the gate,
// so the WG owns matching i/f/g/o columns and applies the cell update via an LDS
// Z-exchange. Double-buffered pipeline with sched fences so chunk kc+1's 12 loads
// stay in flight behind chunk kc's 8 WMMAs.

__global__ __launch_bounds__(256, 1) void lstm_step_k(
    const _Float16* __restrict__ xv,   // [B_][DX]
    const _Float16* __restrict__ xsp,  // [B_][DS]  (plane for this coarse step)
    const _Float16* __restrict__ hin,  // [B_][UNITS]
    const _Float16* __restrict__ WcatT,// [NZ][KTOT]
    const float*    __restrict__ bias, // [NZ]
    float*          __restrict__ c,    // [B_][UNITS]
    _Float16*       __restrict__ hout) // [B_][UNITS]
{
    __shared__ float zl[64 * 256];     // 64 rows x (i|f|g|o x 64 units) = 64 KB

    const int tid   = threadIdx.x;
    const int lane  = tid & 31;
    const int wave  = tid >> 5;
    const int nl    = lane & 15;
    const int hiL   = lane >> 4;
    const int mhalf = wave & 1;        // which 32-row half
    const int g     = wave >> 1;       // gate index 0..3
    const int bm    = blockIdx.x;
    const int bu    = blockIdx.y;
    const int akb   = hiL ? 8 : 0;

    int rowb[2];
    rowb[0] = bm * 64 + mhalf * 32 + nl;
    rowb[1] = rowb[0] + 16;

    floatx8 acc[2][4];
    const _Float16* bptr[4];
#pragma unroll
    for (int j = 0; j < 4; ++j) {
        int col  = g * UNITS + bu * 64 + j * 16 + nl;   // global z column
        float bv = bias[col];
#pragma unroll
        for (int r = 0; r < 8; ++r) { acc[0][j][r] = bv; acc[1][j][r] = bv; }
        bptr[j] = WcatT + (long long)col * KTOT + (hiL ? 8 : 0);
    }

    Frag16 a[2][2], bf[2][4];
    // preload chunk 0
#pragma unroll
    for (int mt = 0; mt < 2; ++mt) {
        const _Float16* ap = a_src(0, rowb[mt], xv, xsp, hin) + akb;
        a[0][mt].h[0] = *(const half8*)(ap);
        a[0][mt].h[1] = *(const half8*)(ap + 16);
    }
#pragma unroll
    for (int j = 0; j < 4; ++j) {
        const _Float16* bp = bptr[j];
        bf[0][j].h[0] = *(const half8*)(bp);
        bf[0][j].h[1] = *(const half8*)(bp + 16);
    }

#pragma unroll
    for (int kc = 0; kc < NKC; ++kc) {
        const int cur = kc & 1, nxt = cur ^ 1;
        if (kc + 1 < NKC) {
            const int kg2 = (kc + 1) * 32;
#pragma unroll
            for (int mt = 0; mt < 2; ++mt) {
                const _Float16* ap = a_src(kg2, rowb[mt], xv, xsp, hin) + akb;
                a[nxt][mt].h[0] = *(const half8*)(ap);
                a[nxt][mt].h[1] = *(const half8*)(ap + 16);
            }
#pragma unroll
            for (int j = 0; j < 4; ++j) {
                const _Float16* bp = bptr[j] + kg2;
                bf[nxt][j].h[0] = *(const half8*)(bp);
                bf[nxt][j].h[1] = *(const half8*)(bp + 16);
            }
        }
        SCHED_FENCE();   // keep prefetch loads issued before this chunk's WMMAs
#pragma unroll
        for (int mt = 0; mt < 2; ++mt)
#pragma unroll
            for (int j = 0; j < 4; ++j)
                acc[mt][j] = __builtin_amdgcn_wmma_f32_16x16x32_f16(
                    false, a[cur][mt].v, false, bf[cur][j].v, (short)0,
                    acc[mt][j], false, false);
        SCHED_FENCE();   // keep WMMAs before next iteration's prefetch
    }

    // Z -> LDS for cross-wave gate exchange
#pragma unroll
    for (int mt = 0; mt < 2; ++mt)
#pragma unroll
        for (int j = 0; j < 4; ++j)
#pragma unroll
            for (int r = 0; r < 8; ++r) {
                int rl = mhalf * 32 + mt * 16 + r + hiL * 8;
                int cl = g * 64 + j * 16 + nl;
                zl[rl * 256 + cl] = acc[mt][j][r];
            }
    __syncthreads();

    // LSTM cell update: 64 rows x 64 units, 16 elems/thread
    for (int e = tid; e < 64 * 64; e += 256) {
        int r = e >> 6, u = e & 63;
        float zi = zl[r * 256 + u];
        float zf = zl[r * 256 + 64 + u];
        float zg = zl[r * 256 + 128 + u];
        float zo = zl[r * 256 + 192 + u];
        long long ci = (long long)(bm * 64 + r) * UNITS + bu * 64 + u;
        float cv = c[ci];
        cv = sigf(zf) * cv + sigf(zi) * tanhf(zg);
        float hv = sigf(zo) * tanhf(cv);
        c[ci]    = cv;
        hout[ci] = (_Float16)hv;
    }
}

// ---------------- dense head: y[t] = clip((h @ Wd + bd) * y_std + y_mean) ----------------
// grid(2), 256 threads; wave w handles rows blockIdx.x*128 + w*16, all 64 cols.
// Same double-buffered, fence-pinned pipeline.

__global__ __launch_bounds__(256, 1) void head_k(
    const _Float16* __restrict__ h,    // [B_][UNITS]
    const _Float16* __restrict__ WdT,  // [NOUT][UNITS]
    const float* __restrict__ bd, const float* __restrict__ ym,
    const float* __restrict__ ys, const float* __restrict__ ymin,
    const float* __restrict__ ymax, float* __restrict__ out, int t)
{
    const int tid  = threadIdx.x;
    const int lane = tid & 31;
    const int wave = tid >> 5;
    const int nl   = lane & 15;
    const int hiL  = lane >> 4;
    const int rowbase = blockIdx.x * 128 + wave * 16;
    const int akb = hiL ? 8 : 0;
    const int bkb = hiL ? 8 : 0;

    const _Float16* arow = h + (long long)(rowbase + nl) * UNITS + akb;
    const _Float16* brow[4];
#pragma unroll
    for (int j = 0; j < 4; ++j)
        brow[j] = WdT + (long long)(j * 16 + nl) * UNITS + bkb;

    floatx8 acc[4];
#pragma unroll
    for (int j = 0; j < 4; ++j) {
        float bv = bd[j * 16 + nl];
#pragma unroll
        for (int r = 0; r < 8; ++r) acc[j][r] = bv;
    }

    Frag16 a[2], bf[2][4];
    a[0].h[0] = *(const half8*)(arow);
    a[0].h[1] = *(const half8*)(arow + 16);
#pragma unroll
    for (int j = 0; j < 4; ++j) {
        bf[0][j].h[0] = *(const half8*)(brow[j]);
        bf[0][j].h[1] = *(const half8*)(brow[j] + 16);
    }

#pragma unroll
    for (int kc = 0; kc < UNITS / 32; ++kc) {
        const int cur = kc & 1, nxt = cur ^ 1;
        if (kc + 1 < UNITS / 32) {
            const int kg2 = (kc + 1) * 32;
            a[nxt].h[0] = *(const half8*)(arow + kg2);
            a[nxt].h[1] = *(const half8*)(arow + kg2 + 16);
#pragma unroll
            for (int j = 0; j < 4; ++j) {
                bf[nxt][j].h[0] = *(const half8*)(brow[j] + kg2);
                bf[nxt][j].h[1] = *(const half8*)(brow[j] + kg2 + 16);
            }
        }
        SCHED_FENCE();
#pragma unroll
        for (int j = 0; j < 4; ++j)
            acc[j] = __builtin_amdgcn_wmma_f32_16x16x32_f16(
                false, a[cur].v, false, bf[cur][j].v, (short)0, acc[j], false, false);
        SCHED_FENCE();
    }

    float lo = *ymin, hc = *ymax;
#pragma unroll
    for (int j = 0; j < 4; ++j) {
        int col = j * 16 + nl;
        float sy = ys[col], my = ym[col];
#pragma unroll
        for (int r = 0; r < 8; ++r) {
            int row = rowbase + r + hiL * 8;
            float v = acc[j][r] * sy + my;
            v = fminf(fmaxf(v, lo), hc);
            out[((long long)row * T_ + t) * NOUT + col] = v;
        }
    }
}

// ---------------- host side ----------------

extern "C" void kernel_launch(void* const* d_in, const int* in_sizes, int n_in,
                              void* d_out, int out_size, void* d_ws, size_t ws_size,
                              hipStream_t stream)
{
    const float* x       = (const float*)d_in[0];
    const float* x_seq   = (const float*)d_in[1];
    const float* x_mean  = (const float*)d_in[2];
    const float* x_std   = (const float*)d_in[3];
    const float* xs_mean = (const float*)d_in[4];
    const float* xs_std  = (const float*)d_in[5];
    const float* Wx      = (const float*)d_in[6];
    const float* Wh      = (const float*)d_in[7];
    const float* bias    = (const float*)d_in[8];
    const float* Wd      = (const float*)d_in[9];
    const float* bd      = (const float*)d_in[10];
    const float* y_mean  = (const float*)d_in[11];
    const float* y_std   = (const float*)d_in[12];
    const float* y_min   = (const float*)d_in[13];
    const float* y_max   = (const float*)d_in[14];
    float* out = (float*)d_out;
    (void)in_sizes; (void)n_in; (void)out_size; (void)ws_size;

    char* ws = (char*)d_ws;
    size_t off = 0;
    auto alloc = [&](size_t bytes) -> char* {
        char* p = ws + off;
        off += (bytes + 255) & ~(size_t)255;
        return p;
    };
    _Float16* xv    = (_Float16*)alloc((size_t)B_ * DX * 2);
    _Float16* xsf   = (_Float16*)alloc((size_t)TSEQ * B_ * DS * 2);
    _Float16* WcatT = (_Float16*)alloc((size_t)NZ * KTOT * 2);
    _Float16* WdT   = (_Float16*)alloc((size_t)NOUT * UNITS * 2);
    float*    cst   = (float*)   alloc((size_t)B_ * UNITS * 4);
    _Float16* h0    = (_Float16*)alloc((size_t)B_ * UNITS * 2);
    _Float16* h1    = (_Float16*)alloc((size_t)B_ * UNITS * 2);

    prep_static_k<<<(B_ * DX) / 256, 256, 0, stream>>>(x, x_mean, x_std, xv);
    prep_seq_k<<<(TSEQ * B_ * DS) / 256, 256, 0, stream>>>(x_seq, xs_mean, xs_std, xsf);
    prep_wcat_k<<<(NZ * KTOT) / 256, 256, 0, stream>>>(Wx, Wh, WcatT);
    prep_wd_k<<<(NOUT * UNITS) / 256, 256, 0, stream>>>(Wd, WdT);
    init_state_k<<<(B_ * UNITS) / 256, 256, 0, stream>>>(cst, h0);

    for (int t = 0; t < T_; ++t) {
        const _Float16* hin  = (t & 1) ? h1 : h0;
        _Float16*       hout = (t & 1) ? h0 : h1;
        const _Float16* xsp  = xsf + (size_t)(t / REP_) * B_ * DS;
        lstm_step_k<<<dim3(4, 8), 256, 0, stream>>>(xv, xsp, hin, WcatT, bias, cst, hout);
        head_k<<<2, 256, 0, stream>>>(hout, WdT, bd, y_mean, y_std, y_min, y_max, out, t);
    }
}
